// TopKCoCN_76459007803896
// MI455X (gfx1250) — matrix operand order repaired and checked by reading, not compiled, not run on hardware
//
#include <hip/hip_runtime.h>
#include <math.h>

// ---------------------------------------------------------------------------
// MI455X (gfx1250) implementation.
//  - wave32, WMMA f32<=f16 16x16x32 for both batched GEMMs.
//  - All matrix operands pre-packed into WMMA VGPR fragment layouts (GEMM1 in
//    ISA 7.12.2 k order; GEMM2: k = 16*(lane>>4) + slot) so GEMM inner loops are b128 loads + v_wmma only.
// ---------------------------------------------------------------------------

typedef __attribute__((ext_vector_type(16))) _Float16 v16h;
typedef __attribute__((ext_vector_type(8)))  float    v8f;

#define Ncnt 100000
#define Ecnt 1000000
#define Dd   128
#define Hh   4
#define Jj   8
#define Kk   1000
#define Cc   40
#define Gg   32            // J*H groups
#define M64  131072        // 2^17 padded sort size per head
#define MT   64            // row tiles of 16  (K padded to 1024)
#define NT   8             // col tiles of 16  (D = 128)
#define KT1  4             // k tiles of 32 for GEMM1 (D = 128)
#define KT2  32            // k tiles of 32 for GEMM2 (K padded to 1024)
#define TOPK (Jj*Kk)       // 8000 per head

// ---------------- graph aggregation -> scores ------------------------------

__global__ void k_deg_init(float* deg) {
    int n = blockIdx.x * blockDim.x + threadIdx.x;
    if (n < Ncnt) deg[n] = 1.0f;
}

__global__ void k_deg_acc(const int* ei, float* deg) {
    int e = blockIdx.x * blockDim.x + threadIdx.x;
    if (e < Ecnt) atomicAdd(&deg[ei[Ecnt + e]], 1.0f);
}

// p[n][h] = x[n] . w_rank[:,h]
__global__ void k_proj(const float* x, const float* wr, float* p) {
    int t = blockIdx.x * blockDim.x + threadIdx.x;
    if (t >= Ncnt * Hh) return;
    int n = t >> 2, h = t & 3;
    const float* xr = x + (size_t)n * Dd;
    float s = 0.f;
    for (int d = 0; d < Dd; ++d) s += xr[d] * wr[d * Hh + h];
    p[t] = s;
}

// self term: scores[n][h] = p[n][h] / deg[n]
__global__ void k_score_self(const float* p, const float* deg, float* sc) {
    int t = blockIdx.x * blockDim.x + threadIdx.x;
    if (t < Ncnt * Hh) sc[t] = p[t] / deg[t >> 2];
}

// edge term: scores[v][h] += rsqrt(deg[u]*deg[v]) * p[u][h]
__global__ void k_score_edge(const int* ei, const float* p, const float* deg, float* sc) {
    int e = blockIdx.x * blockDim.x + threadIdx.x;
    if (e >= Ecnt) return;
    int u = ei[e], v = ei[Ecnt + e];
    float nrm = rsqrtf(deg[u] * deg[v]);
    #pragma unroll
    for (int h = 0; h < Hh; ++h)
        atomicAdd(&sc[v * Hh + h], nrm * p[u * Hh + h]);
}

// ---------------- exact top-k via padded bitonic sort ----------------------
// key = orderable(score) << 32 | ~index  (descending sort == jax top_k order)

__global__ void k_keys(const float* sc, unsigned long long* keys) {
    unsigned t = blockIdx.x * blockDim.x + threadIdx.x;
    if (t >= Hh * M64) return;
    unsigned h = t >> 17, i = t & (M64 - 1);
    unsigned long long kv = 0ull;
    if (i < Ncnt) {
        unsigned u = __float_as_uint(sc[(size_t)i * Hh + h]);
        u = (u & 0x80000000u) ? ~u : (u | 0x80000000u);
        kv = ((unsigned long long)u << 32) | (unsigned)(~i);
    }
    keys[t] = kv;
}

__global__ void k_bitonic(unsigned long long* keys, unsigned kk, unsigned jj) {
    unsigned t = blockIdx.x * blockDim.x + threadIdx.x;
    if (t >= Hh * M64) return;
    unsigned h = t >> 17, i = t & (M64 - 1);
    unsigned ixj = i ^ jj;
    if (ixj <= i) return;
    unsigned long long* kb = keys + ((size_t)h << 17);
    unsigned long long a = kb[i], b = kb[ixj];
    bool up = ((i & kk) == 0);           // descending overall
    if (up ? (a < b) : (a > b)) { kb[i] = b; kb[ixj] = a; }
}

__global__ void k_extract(const unsigned long long* keys, const float* sc,
                          int* sidx, float* vals) {
    int t = blockIdx.x * blockDim.x + threadIdx.x;
    if (t >= Hh * TOPK) return;
    int h = t / TOPK, pos = t - h * TOPK;
    unsigned long long kv = keys[((size_t)h << 17) + pos];
    int idx = (int)(~(unsigned)kv);
    sidx[t] = idx;
    vals[t] = sc[(size_t)idx * Hh + h];
}

// ---------------- membership map + packed operand construction -------------

__global__ void k_buildM(const int* sidx, int* Mmap) {
    int t = blockIdx.x * blockDim.x + threadIdx.x;
    if (t >= Gg * Kk) return;
    int g = t / Kk, k = t - g * Kk;
    int jj = g >> 2, h = g & 3;                       // g = jj*H + h
    int idx = sidx[h * TOPK + jj * Kk + k];
    Mmap[(size_t)g * Ncnt + idx] = k + 1;
}

// Gather node features straight into WMMA A-fragment layout (f16).
// Fragment (g, mt, kt, lane): lane holds row m=lane&15; K runs
// [hs*8, hs*8+8) and [16+hs*8, 16+hs*8+8) with hs = lane>>4 (ISA 7.12.2).
__global__ void k_gather(const float* x, const int* sidx, _Float16* A1) {
    unsigned t = blockIdx.x * blockDim.x + threadIdx.x;   // 2^18 threads
    int lane = t & 31;
    int kt   = (t >> 5) & 3;
    int mt   = (t >> 7) & 63;
    int g    = t >> 13;
    int m = lane & 15, hs = lane >> 4;
    int row = mt * 16 + m;
    v16h vv;
    if (row < Kk) {
        int jj = g >> 2, h = g & 3;
        int node = sidx[h * TOPK + jj * Kk + row];
        const float* xr = x + (size_t)node * Dd + kt * 32 + hs * 8;
        #pragma unroll
        for (int s = 0; s < 8; ++s) vv[s] = (_Float16)xr[s];
        #pragma unroll
        for (int s = 0; s < 8; ++s) vv[8 + s] = (_Float16)xr[16 + s];
    } else {
        #pragma unroll
        for (int s = 0; s < 16; ++s) vv[s] = (_Float16)0.f;
    }
    ((v16h*)A1)[(((size_t)g * MT + mt) * KT1 + kt) * 32 + lane] = vv;
}

// Pack W1 (D x D, f32 row-major) into WMMA B-fragment layout:
// lane: n = lane&15, ksel = lane>>4; slot s -> K = ksel*8 + s (s<8), 16 + ksel*8 + (s-8) (s>=8).
__global__ void k_packW1(const float* W1, _Float16* W1p) {
    int t = blockIdx.x * blockDim.x + threadIdx.x;    // 1024 threads
    if (t >= KT1 * NT * 32) return;
    int lane = t & 31, nt = (t >> 5) & 7, kt = t >> 8;
    int n = lane & 15, ksel = lane >> 4;
    v16h vv;
    #pragma unroll
    for (int s = 0; s < 16; ++s) {
        int krow = kt * 32 + (s < 8 ? ksel * 8 + s : 16 + ksel * 8 + (s - 8));
        vv[s] = (_Float16)W1[(size_t)krow * Dd + nt * 16 + n];
    }
    ((v16h*)W1p)[(size_t)(kt * NT + nt) * 32 + lane] = vv;
}

// Scatter edge weights directly into adj A-fragment layout (f16).
__global__ void k_adj(const int* ei, const float* ea, const int* Mmap, _Float16* adjP) {
    int e = blockIdx.x * blockDim.x + threadIdx.x;
    if (e >= Ecnt) return;
    int u = ei[e], v = ei[Ecnt + e];
    _Float16 hw = (_Float16)ea[e];
    for (int g = 0; g < Gg; ++g) {
        int lu = Mmap[(size_t)g * Ncnt + u];
        int lv = Mmap[(size_t)g * Ncnt + v];
        if (lu > 0 && lv > 0) {
            int row = lu - 1, col = lv - 1;
            int mt = row >> 4, m = row & 15;
            int kt2 = col >> 5, kk = col & 31;
            int hs = kk >> 4;
            int slot = kk & 15;
            int lane = hs * 16 + m;
            adjP[((((size_t)g * MT + mt) * KT2 + kt2) * 32 + lane) * 16 + slot] = hw;
        }
    }
}

// ---------------- GEMM1: h1 = feats @ W1, epilogue writes B-frag for GEMM2 -

__global__ void k_gemm1(const _Float16* A1, const _Float16* W1p, _Float16* h1B) {
    int wid  = blockIdx.x * 8 + (threadIdx.x >> 5);     // 16384 waves
    int lane = threadIdx.x & 31;
    int nt = wid & 7, mt = (wid >> 3) & 63, g = wid >> 9;
    const v16h* Af = (const v16h*)A1 + (((size_t)g * MT + mt) * KT1) * 32 + lane;
    const v16h* Bf = (const v16h*)W1p + lane;
    v8f acc = {};
    #pragma unroll
    for (int kt = 0; kt < KT1; ++kt) {
        v16h a = Af[(size_t)kt * 32];
        v16h b = Bf[(size_t)(kt * NT + nt) * 32];
        acc = __builtin_amdgcn_wmma_f32_16x16x32_f16(false, a, false, b,
                                                     (short)0, acc, false, false);
    }
    // Output tile rows = K-dim of GEMM2. Write as f16 into B-fragment layout:
    // 16 rows share (kt2 = mt>>1, ksel = mt&1); lane' = ksel*16 + n, slot = M.
    int n = lane & 15, msel = lane >> 4;
    int kt2 = mt >> 1, ksel = mt & 1;
    size_t fi = (((size_t)g * KT2 + kt2) * NT + nt) * 32 + (ksel * 16 + n);
    alignas(16) _Float16 hv[8];
    #pragma unroll
    for (int r = 0; r < 8; ++r) hv[r] = (_Float16)acc[r];
    *reinterpret_cast<uint4*>((_Float16*)h1B + fi * 16 + msel * 8) =
        *reinterpret_cast<uint4*>(hv);
}

// ---------------- GEMM2: msg = adj @ h1 + gated pooling epilogue -----------

__global__ void k_gemm2(const _Float16* adjP, const _Float16* h1B,
                        const float* vals, float* pooled_part) {
    int wid  = blockIdx.x * 8 + (threadIdx.x >> 5);     // 16384 waves
    int lane = threadIdx.x & 31;
    int nt = wid & 7, mt = (wid >> 3) & 63, g = wid >> 9;
    const v16h* Af = (const v16h*)adjP + (((size_t)g * MT + mt) * KT2) * 32 + lane;
    const v16h* Bf = (const v16h*)h1B + ((size_t)g * KT2 * NT) * 32 + lane;
    v8f acc = {};
    for (int kt2 = 0; kt2 < KT2; ++kt2) {
        v16h a = Af[(size_t)kt2 * 32];
        v16h b = Bf[(size_t)(kt2 * NT + nt) * 32];
        acc = __builtin_amdgcn_wmma_f32_16x16x32_f16(false, a, false, b,
                                                     (short)0, acc, false, false);
    }
    int n = lane & 15, msel = lane >> 4;
    int d = nt * 16 + n;
    int jj = g >> 2, h = g & 3;
    float sum = 0.f;
    #pragma unroll
    for (int r = 0; r < 8; ++r) {
        int k = mt * 16 + msel * 8 + r;
        if (k < Kk) {
            float v = acc[r];
            v = v > 0.f ? v : 0.f;                          // relu
            float s = vals[h * TOPK + jj * Kk + k];         // perm_in[jj,h,k]
            sum += v * (1.f / (1.f + __expf(-s)));          // * sigmoid
        }
    }
    atomicAdd(&pooled_part[(size_t)g * Dd + d], sum);
}

// ---------------- final: mean pool, 8x128x40 GEMM, log_softmax -------------

__global__ void k_final(const float* pooled_part, const float* W2, float* out) {
    __shared__ float pooled[Jj][Dd];
    __shared__ float o[Jj][Cc];
    int t = threadIdx.x;                      // block of 1024
    if (t < Jj * Dd) {
        int jj = t >> 7, d = t & 127;
        float s = 0.f;
        #pragma unroll
        for (int h = 0; h < Hh; ++h) s += pooled_part[(size_t)(jj * Hh + h) * Dd + d];
        pooled[jj][d] = s / (float)(Hh * Kk);
    }
    __syncthreads();
    if (t < Jj * Cc) {
        int jj = t / Cc, c = t - jj * Cc;
        float s = 0.f;
        for (int d = 0; d < Dd; ++d) s += pooled[jj][d] * W2[(size_t)d * Cc + c];
        o[jj][c] = s;
    }
    __syncthreads();
    if (t < Jj) {
        float mx = -1e30f;
        for (int c = 0; c < Cc; ++c) mx = fmaxf(mx, o[t][c]);
        float se = 0.f;
        for (int c = 0; c < Cc; ++c) se += __expf(o[t][c] - mx);
        float lse = mx + __logf(se);
        for (int c = 0; c < Cc; ++c) out[t * Cc + c] = o[t][c] - lse;
    }
}

// ---------------------------------------------------------------------------

extern "C" void kernel_launch(void* const* d_in, const int* in_sizes, int n_in,
                              void* d_out, int out_size, void* d_ws, size_t ws_size,
                              hipStream_t stream) {
    (void)in_sizes; (void)n_in; (void)out_size; (void)ws_size;
    const float* x   = (const float*)d_in[0];   // N x D
    const float* ea  = (const float*)d_in[1];   // E
    const float* wr  = (const float*)d_in[2];   // D x H
    const float* W1  = (const float*)d_in[3];   // D x D
    const float* W2  = (const float*)d_in[4];   // D x C
    const int*   ei  = (const int*)d_in[5];     // 2 x E
    float* out = (float*)d_out;                 // J x C

    char* ws = (char*)d_ws;
    size_t off = 0;
    auto take = [&](size_t bytes) {
        char* ptr = ws + off;
        off += (bytes + 255) & ~(size_t)255;
        return ptr;
    };
    float* deg  = (float*)take((size_t)Ncnt * 4);
    float* p    = (float*)take((size_t)Ncnt * Hh * 4);
    float* sc   = (float*)take((size_t)Ncnt * Hh * 4);
    unsigned long long* keys = (unsigned long long*)take((size_t)Hh * M64 * 8);
    int*   sidx = (int*)take((size_t)Hh * TOPK * 4);
    float* vals = (float*)take((size_t)Hh * TOPK * 4);
    int*   Mmap = (int*)take((size_t)Gg * Ncnt * 4);
    _Float16* A1   = (_Float16*)take((size_t)Gg * MT * KT1 * 32 * 16 * 2);   //  8.4 MB
    _Float16* W1p  = (_Float16*)take((size_t)KT1 * NT * 32 * 16 * 2);
    _Float16* h1B  = (_Float16*)take((size_t)Gg * KT2 * NT * 32 * 16 * 2);   //  8.4 MB
    _Float16* adjP = (_Float16*)take((size_t)Gg * MT * KT2 * 32 * 16 * 2);   // 67.1 MB
    float* pooled_part = (float*)take((size_t)Gg * Dd * 4);

    const int B = 256;
    // zero-fill scatter targets (graph-capturable memset nodes)
    hipMemsetAsync(Mmap, 0, (size_t)Gg * Ncnt * 4, stream);
    hipMemsetAsync(adjP, 0, (size_t)Gg * MT * KT2 * 32 * 16 * 2, stream);
    hipMemsetAsync(pooled_part, 0, (size_t)Gg * Dd * 4, stream);

    k_deg_init  <<<(Ncnt + B - 1) / B, B, 0, stream>>>(deg);
    k_deg_acc   <<<(Ecnt + B - 1) / B, B, 0, stream>>>(ei, deg);
    k_proj      <<<(Ncnt * Hh + B - 1) / B, B, 0, stream>>>(x, wr, p);
    k_score_self<<<(Ncnt * Hh + B - 1) / B, B, 0, stream>>>(p, deg, sc);
    k_score_edge<<<(Ecnt + B - 1) / B, B, 0, stream>>>(ei, p, deg, sc);

    k_keys<<<(Hh * M64) / B, B, 0, stream>>>(sc, keys);
    for (unsigned kk = 2; kk <= M64; kk <<= 1)
        for (unsigned jj = kk >> 1; jj > 0; jj >>= 1)
            k_bitonic<<<(Hh * M64) / B, B, 0, stream>>>(keys, kk, jj);
    k_extract<<<(Hh * TOPK + B - 1) / B, B, 0, stream>>>(keys, sc, sidx, vals);

    k_buildM<<<(Gg * Kk + B - 1) / B, B, 0, stream>>>(sidx, Mmap);
    k_gather<<<(Gg * MT * KT1 * 32) / B, B, 0, stream>>>(x, sidx, A1);
    k_packW1<<<(KT1 * NT * 32) / B, B, 0, stream>>>(W1, W1p);
    k_adj   <<<(Ecnt + B - 1) / B, B, 0, stream>>>(ei, ea, Mmap, adjP);

    k_gemm1<<<(Gg * MT * NT) / 8, B, 0, stream>>>(A1, W1p, h1B);
    k_gemm2<<<(Gg * MT * NT) / 8, B, 0, stream>>>(adjP, h1B, vals, pooled_part);

    k_final<<<1, 1024, 0, stream>>>(pooled_part, W2, out);
}
